// BertAttention_78735340470733
// MI455X (gfx1250) — compile-verified, matrix operand order checked by reading, not run on hardware
//
#include <hip/hip_runtime.h>

typedef _Float16 f16;
typedef _Float16 v8h  __attribute__((ext_vector_type(8)));
typedef _Float16 v16h __attribute__((ext_vector_type(16)));
typedef float    v8f  __attribute__((ext_vector_type(8)));

#define HIDDEN 768
#define HEADS  12
#define HD     64
#define SEQ    2048
#define BATCH  2

#define WMMA_F16(A, B, C) \
  __builtin_amdgcn_wmma_f32_16x16x32_f16(false, (A), false, (B), (short)0, (C), false, false)

static __device__ __forceinline__ v16h cat8(v8h lo, v8h hi) {
  return __builtin_shufflevector(lo, hi, 0,1,2,3,4,5,6,7,8,9,10,11,12,13,14,15);
}

// ---------------------------------------------------------------- f32 -> f16
__global__ void cvt_f32_f16(const float* __restrict__ in, f16* __restrict__ out, int n) {
  int i = blockIdx.x * blockDim.x + threadIdx.x;
  int stride = gridDim.x * blockDim.x;
  for (; i < n; i += stride) out[i] = (f16)in[i];
}

// ---------------------------------------------------------------- projection
// Y = X(4096x768) @ W^T(768x768) + bias. Each wave computes a 16x64 strip
// (one A-frag reused across 4 N-tiles -> 10 loads : 4 WMMAs per k-step).
// flavor 0: Q -> [B,H,S,64] (scaled) | 1: K -> [B,H,S,64] | 2: V -> Vt[B,H,64,S]
__global__ __launch_bounds__(128)
void proj_kernel(const f16* __restrict__ X, const f16* __restrict__ W,
                 const float* __restrict__ bias, f16* __restrict__ out,
                 int flavor, float scale) {
  const int lane = threadIdx.x & 31;
  const int wave = threadIdx.x >> 5;
  const int lr   = lane & 15;
  const int hf   = lane >> 4;
  const int kb0  = hf * 8;

  const int m0 = (blockIdx.x * 4 + wave) * 16;   // row tile (0..4095)
  const int h  = blockIdx.y;                     // head == 64-col group

  const f16* xrow = X + (size_t)(m0 + lr) * HIDDEN;
  const f16* wrow[4];
  #pragma unroll
  for (int t = 0; t < 4; ++t)
    wrow[t] = W + (size_t)(h * 64 + t * 16 + lr) * HIDDEN;

  v8f acc[4] = {};
  #pragma unroll
  for (int ks = 0; ks < HIDDEN; ks += 32) {
    v16h a = cat8(*(const v8h*)(xrow + ks + kb0),
                  *(const v8h*)(xrow + ks + 16 + kb0));
    #pragma unroll
    for (int t = 0; t < 4; ++t) {
      v16h bfr = cat8(*(const v8h*)(wrow[t] + ks + kb0),
                      *(const v8h*)(wrow[t] + ks + 16 + kb0));
      acc[t] = WMMA_F16(a, bfr, acc[t]);
    }
  }

  #pragma unroll
  for (int t = 0; t < 4; ++t) {
    const int d  = t * 16 + lr;
    const float bv = bias[h * 64 + d];
    #pragma unroll
    for (int r = 0; r < 8; ++r) {
      const int row = m0 + r + 8 * hf;
      const int b   = row >> 11;                 // SEQ == 2048
      const int s   = row & (SEQ - 1);
      float y = (acc[t][r] + bv) * scale;
      if (flavor == 2) {
        out[(((size_t)(b * HEADS + h) * HD) + d) * SEQ + s] = (f16)y;   // Vt[d,s]
      } else {
        out[(((size_t)(b * HEADS + h) * SEQ) + s) * HD + d] = (f16)y;   // Q/K[s,d]
      }
    }
  }
}

// ---------------------------------------------------------------- attention
// grid: (SEQ/128, B*H), block 128 (4 waves). Wave owns 32 query rows (two
// 16-row tiles), streams keys in chunks of 32 with online softmax. K/V frags
// loaded once per chunk and reused by both query tiles -> 1:1 VMEM:WMMA.
__global__ __launch_bounds__(128)
void attn_kernel(const f16* __restrict__ Q, const f16* __restrict__ K,
                 const f16* __restrict__ Vt, const float* __restrict__ mask,
                 float* __restrict__ out) {
  __shared__ float smask[SEQ];
  const int bh = blockIdx.y;
  const int b  = bh / HEADS;
  const int h  = bh % HEADS;

  for (int i = threadIdx.x; i < SEQ; i += blockDim.x)
    smask[i] = mask[(size_t)b * SEQ + i];
  __syncthreads();

  const int lane = threadIdx.x & 31;
  const int wave = threadIdx.x >> 5;
  const int lr   = lane & 15;
  const int hf   = lane >> 4;
  const int kb0  = hf * 8;
  const int q0   = blockIdx.x * 128 + wave * 32;   // 32-query strip base

  // Q as B-operand of S^T = K x Q^T (two query tiles, resident)
  v16h Qb[2][2];
  #pragma unroll
  for (int u = 0; u < 2; ++u) {
    const f16* qrow = Q + ((size_t)bh * SEQ + q0 + u * 16 + lr) * HD;
    #pragma unroll
    for (int ks = 0; ks < 2; ++ks)
      Qb[u][ks] = cat8(*(const v8h*)(qrow + ks * 32 + kb0),
                       *(const v8h*)(qrow + ks * 32 + 16 + kb0));
  }

  const f16* Kbh = K  + (size_t)bh * SEQ * HD;
  const f16* Vbh = Vt + (size_t)bh * HD * SEQ;

  v8f O[2][4] = {};
  float m_run[2] = {-1e30f, -1e30f};
  float l_run[2] = {0.f, 0.f};

  for (int kb = 0; kb < SEQ; kb += 32) {
    // ---- shared K A-frags (2 key tiles x 2 k-steps) and V B-frags (4 d-tiles)
    v16h ka[2][2];
    #pragma unroll
    for (int c = 0; c < 2; ++c)
      #pragma unroll
      for (int ks = 0; ks < 2; ++ks) {
        const f16* kr = Kbh + (size_t)(kb + c * 16 + lr) * HD + ks * 32 + kb0;
        ka[c][ks] = cat8(*(const v8h*)kr, *(const v8h*)(kr + 16));
      }
    v16h vb[4];
    #pragma unroll
    for (int t = 0; t < 4; ++t) {
      const f16* vr = Vbh + (size_t)(t * 16 + lr) * SEQ + kb + kb0;
      vb[t] = cat8(*(const v8h*)vr, *(const v8h*)(vr + 16));
    }
    float mk0[8], mk1[8];
    #pragma unroll
    for (int r = 0; r < 8; ++r) {
      mk0[r] = smask[kb + r + 8 * hf];
      mk1[r] = smask[kb + 16 + r + 8 * hf];
    }

    if (kb + 32 < SEQ) {   // gfx1250 global_prefetch_b8 for next chunk
      __builtin_prefetch(Kbh + (size_t)(kb + 32 + lr) * HD, 0, 1);
      __builtin_prefetch(Vbh + (size_t)lr * SEQ + kb + 32, 0, 1);
    }

    #pragma unroll
    for (int u = 0; u < 2; ++u) {
      // ---- S^T tiles for query tile u
      v8f c0 = {}, c1 = {};
      #pragma unroll
      for (int ks = 0; ks < 2; ++ks) {
        c0 = WMMA_F16(ka[0][ks], Qb[u][ks], c0);
        c1 = WMMA_F16(ka[1][ks], Qb[u][ks], c1);
      }

      // ---- online softmax; lane carries query lr (replicated in halves)
      float p[16];
      float mx = -1e30f;
      #pragma unroll
      for (int r = 0; r < 8; ++r) {
        float s0 = c0[r] + mk0[r];
        float s1 = c1[r] + mk1[r];
        p[r] = s0; p[8 + r] = s1;
        mx = fmaxf(mx, fmaxf(s0, s1));
      }
      mx = fmaxf(mx, __shfl_xor(mx, 16, 32));
      const float m_new = fmaxf(m_run[u], mx);
      const float alpha = __expf(m_run[u] - m_new);

      v16h pa;              // C-layout of S^T tiles == A-layout of P
      float lsum = 0.f;
      #pragma unroll
      for (int r = 0; r < 8; ++r) {
        float e0 = __expf(p[r] - m_new);
        float e1 = __expf(p[8 + r] - m_new);
        lsum += e0 + e1;
        pa[r] = (f16)e0; pa[8 + r] = (f16)e1;
      }
      lsum += __shfl_xor(lsum, 16, 32);
      l_run[u] = l_run[u] * alpha + lsum;
      m_run[u] = m_new;

      // ---- rescale O rows (row M = r + 8*hf; alpha for query j lives in lane j)
      #pragma unroll
      for (int r = 0; r < 8; ++r) {
        float ar = __shfl(alpha, r + 8 * hf, 32);
        #pragma unroll
        for (int t = 0; t < 4; ++t) O[u][t][r] *= ar;
      }

      // ---- O += P x Vchunk (V frags shared across u)
      #pragma unroll
      for (int t = 0; t < 4; ++t)
        O[u][t] = WMMA_F16(pa, vb[t], O[u][t]);
    }
  }

  // ---- normalize and store [B,S,768] f32
  #pragma unroll
  for (int u = 0; u < 2; ++u) {
    const float inv = 1.0f / l_run[u];
    #pragma unroll
    for (int r = 0; r < 8; ++r) {
      float ir = __shfl(inv, r + 8 * hf, 32);
      const int s = q0 + u * 16 + r + 8 * hf;
      #pragma unroll
      for (int t = 0; t < 4; ++t)
        out[((size_t)b * SEQ + s) * HIDDEN + h * 64 + t * 16 + lr] = O[u][t][r] * ir;
    }
  }
}

// ---------------------------------------------------------------- launcher
extern "C" void kernel_launch(void* const* d_in, const int* in_sizes, int n_in,
                              void* d_out, int out_size, void* d_ws, size_t ws_size,
                              hipStream_t stream) {
  const float* hidden  = (const float*)d_in[0];
  const float* context = (const float*)d_in[1];
  const float* amask   = (const float*)d_in[2];
  const float* Wq = (const float*)d_in[3];
  const float* bq = (const float*)d_in[4];
  const float* Wk = (const float*)d_in[5];
  const float* bk = (const float*)d_in[6];
  const float* Wv = (const float*)d_in[7];
  const float* bv = (const float*)d_in[8];
  float* outp = (float*)d_out;

  const size_t ACT = (size_t)BATCH * SEQ * HIDDEN;   // 3,145,728 elems
  const size_t WSZ = (size_t)HIDDEN * HIDDEN;        //   589,824 elems
  const size_t A_BYTES = ACT * sizeof(f16);          // 6,291,456
  const size_t W_BYTES = WSZ * sizeof(f16);          // 1,179,648

  char* ws = (char*)d_ws;
  f16* Xh16 = (f16*)(ws);
  f16* Xc16 = (f16*)(ws + A_BYTES);
  f16* Wq16 = (f16*)(ws + 2 * A_BYTES);
  f16* Wk16 = (f16*)(ws + 2 * A_BYTES + W_BYTES);
  f16* Wv16 = (f16*)(ws + 2 * A_BYTES + 2 * W_BYTES);
  f16* Q16  = (f16*)(ws + 2 * A_BYTES + 3 * W_BYTES);
  f16* K16  = (f16*)(ws + 3 * A_BYTES + 3 * W_BYTES);
  f16* Vt16 = (f16*)(ws + 4 * A_BYTES + 3 * W_BYTES);

  cvt_f32_f16<<<2048, 256, 0, stream>>>(hidden,  Xh16, (int)ACT);
  cvt_f32_f16<<<2048, 256, 0, stream>>>(context, Xc16, (int)ACT);
  cvt_f32_f16<<<1024, 256, 0, stream>>>(Wq, Wq16, (int)WSZ);
  cvt_f32_f16<<<1024, 256, 0, stream>>>(Wk, Wk16, (int)WSZ);
  cvt_f32_f16<<<1024, 256, 0, stream>>>(Wv, Wv16, (int)WSZ);

  const float qscale = 1.0f / sqrtf((float)HD + 1e-5f);
  dim3 pgrid((BATCH * SEQ) / 64, HEADS);             // 4 waves x 16 rows per block
  proj_kernel<<<pgrid, 128, 0, stream>>>(Xh16, Wq16, bq, Q16,  0, qscale);
  proj_kernel<<<pgrid, 128, 0, stream>>>(Xc16, Wk16, bk, K16,  1, 1.0f);
  proj_kernel<<<pgrid, 128, 0, stream>>>(Xc16, Wv16, bv, Vt16, 2, 1.0f);

  dim3 agrid(SEQ / 128, BATCH * HEADS);
  attn_kernel<<<agrid, 128, 0, stream>>>(Q16, K16, Vt16, amask, outp);
}
